// RGCN_22857815949325
// MI455X (gfx1250) — matrix operand order repaired and checked by reading, not compiled, not run on hardware
//
#include <hip/hip_runtime.h>

// ---------------------------------------------------------------------------
// RGCN (2-layer) for MI455X / gfx1250.
//   Layer: out = relu( x@W_root + b + sum_r mean_{(r,dst)} (x[src] @ W_r) )
// Transform-then-aggregate. Per-relation GEMMs via v_wmma_f32_16x16x32_bf16
// (bf16 in, f32 accumulate, K=128 as 4 chained WMMAs). x_all stored bf16 to
// halve the dominant random-gather HBM traffic (~410 MB/layer). Scatter uses
// one wave per edge with f32 atomics into the L2-resident [N,128] output.
// ---------------------------------------------------------------------------

typedef __attribute__((ext_vector_type(16))) __bf16 v16bf;
typedef __attribute__((ext_vector_type(8)))  float  v8f;
typedef unsigned short ushort_t;

#define HIDDEN 128
#define NREL 8
#define NMAT 9   // 8 relations + root

union V16U {
    v16bf v;
    uint4 q[2];
};

__device__ __forceinline__ ushort_t f32_to_bf16(float f) {
    unsigned int u = __float_as_uint(f);
    u += 0x7FFFu + ((u >> 16) & 1u);   // round to nearest even
    return (ushort_t)(u >> 16);
}
__device__ __forceinline__ float bf16_to_f32(unsigned int s) {
    return __uint_as_float(s << 16);
}

// xb[n][0..127] = bf16( emb[node_idx[n]][0..127] ), 4 features per thread
__global__ void k_cast_x(const int* __restrict__ node_idx,
                         const float* __restrict__ emb,
                         ushort_t* __restrict__ xb, int totalQ) {
    int i = blockIdx.x * blockDim.x + threadIdx.x;
    if (i >= totalQ) return;                  // totalQ = N*32
    int n = i >> 5, fq = (i & 31) << 2;
    const float4 v = *(const float4*)(emb + (size_t)node_idx[n] * HIDDEN + fq);
    ushort4 o;
    o.x = f32_to_bf16(v.x); o.y = f32_to_bf16(v.y);
    o.z = f32_to_bf16(v.z); o.w = f32_to_bf16(v.w);
    *(ushort4*)(xb + (size_t)n * HIDDEN + fq) = o;
}

// Wt[m][f][d] = bf16( m<8 ? W_rel[m][d][f] : W_root[d][f] )  (transpose for B-frag)
__global__ void k_prep_w(const float* __restrict__ wrel,
                         const float* __restrict__ wroot,
                         ushort_t* __restrict__ wt) {
    int i = blockIdx.x * blockDim.x + threadIdx.x;
    if (i >= NMAT * HIDDEN * HIDDEN) return;
    int m = i / (HIDDEN * HIDDEN);
    int r = i - m * (HIDDEN * HIDDEN);
    int f = r >> 7, d = r & 127;
    float v = (m < NREL) ? wrel[(size_t)m * HIDDEN * HIDDEN + (size_t)d * HIDDEN + f]
                         : wroot[(size_t)d * HIDDEN + f];
    wt[i] = f32_to_bf16(v);
}

__global__ void k_zero_i32(int* __restrict__ p, int n) {
    int i = blockIdx.x * blockDim.x + threadIdx.x;
    if (i < n) p[i] = 0;
}

__global__ void k_count_deg(const int* __restrict__ dst,
                            const int* __restrict__ et,
                            int* __restrict__ deg, int E, int N) {
    int e = blockIdx.x * blockDim.x + threadIdx.x;
    if (e >= E) return;
    atomicAdd(&deg[(size_t)et[e] * N + dst[e]], 1);
}

__global__ void k_edge_norm(const int* __restrict__ dst,
                            const int* __restrict__ et,
                            const int* __restrict__ deg,
                            float* __restrict__ norm, int E, int N) {
    int e = blockIdx.x * blockDim.x + threadIdx.x;
    if (e >= E) return;
    float d = (float)deg[(size_t)et[e] * N + dst[e]];
    norm[e] = 1.0f / fmaxf(d, 1.0f);
}

// One 16x16 output tile per wave; K=128 via 4 chained v_wmma_f32_16x16x32_bf16.
// blockIdx.x = row tile, blockIdx.y = matrix (0..7 relations -> bf16 x_all,
// 8 = root -> f32 out + bias). 256 threads = 8 waves covering 8 column tiles.
__global__ void __launch_bounds__(256)
k_gemm(const ushort_t* __restrict__ xb,   // [N][128] bf16
       const ushort_t* __restrict__ wt,   // [9][128][128] bf16 (f-major)
       const float* __restrict__ bias,    // [128]
       ushort_t* __restrict__ xall,       // [8][N][128] bf16
       float* __restrict__ out,           // [N][128] f32
       int N) {
    const int wave = threadIdx.x >> 5;
    const int lane = threadIdx.x & 31;
    const int hi   = lane >> 4;      // lane group 0/1
    const int ln   = lane & 15;
    const int m    = blockIdx.y;
    const int rowbase = blockIdx.x * 16;
    const bool full = (rowbase + 16 <= N);   // block-uniform fast path
    int arow = rowbase + ln; if (arow > N - 1) arow = N - 1;

    const ushort_t* xrow = xb + (size_t)arow * HIDDEN;
    const int col = wave * 16 + ln;
    const ushort_t* wrow = wt + ((size_t)m * HIDDEN + col) * HIDDEN;

    v8f c = {};
#pragma unroll
    for (int t = 0; t < 4; ++t) {
        const int k0 = t * 32;
        V16U a, b;
        // A 16x32 bf16: lanes0-15 rows M=0..15 hold K {k0..k0+7, k0+16..k0+23};
        // lanes16-31 hold K {k0+8..k0+15, k0+24..k0+31}.
        a.q[0] = *(const uint4*)(xrow + k0 + hi * 8);
        a.q[1] = *(const uint4*)(xrow + k0 + 16 + hi * 8);
        // B 32x16 bf16: lane column = lane&15; same K order as A
        b.q[0] = *(const uint4*)(wrow + k0 + hi * 8);
        b.q[1] = *(const uint4*)(wrow + k0 + 16 + hi * 8);
        c = __builtin_amdgcn_wmma_f32_16x16x32_bf16(
                false, a.v, false, b.v, (short)0, c, false, false);
    }

    // D layout: VGPR j -> row = rowbase + j + hi*8, col = colbase + (lane&15)
    const int rbase = rowbase + hi * 8;
    if (m < NREL) {
        ushort_t* orow = xall + (size_t)m * N * HIDDEN + col;
        if (full) {
#pragma unroll
            for (int j = 0; j < 8; ++j)
                orow[(size_t)(rbase + j) * HIDDEN] = f32_to_bf16(c[j]);
        } else {
            for (int j = 0; j < 8; ++j)
                if (rbase + j < N) orow[(size_t)(rbase + j) * HIDDEN] = f32_to_bf16(c[j]);
        }
    } else {
        const float bb = bias[col];
        float* oc = out + col;
        if (full) {
#pragma unroll
            for (int j = 0; j < 8; ++j)
                oc[(size_t)(rbase + j) * HIDDEN] = c[j] + bb;
        } else {
            for (int j = 0; j < 8; ++j)
                if (rbase + j < N) oc[(size_t)(rbase + j) * HIDDEN] = c[j] + bb;
        }
    }
}

// One wave per edge: out[dst] += x_all[et][src][:] * norm[e].
// Edge scalars are wave-uniform (-> SMEM loads); each lane handles 4 features
// via one 8B load + 4 f32 atomics into the L2-resident output.
__global__ void __launch_bounds__(256)
k_scatter(const ushort_t* __restrict__ xall,
          const int* __restrict__ src,
          const int* __restrict__ dst,
          const int* __restrict__ et,
          const float* __restrict__ norm,
          float* __restrict__ out, int E, int N) {
    const int e    = blockIdx.x * 8 + (threadIdx.x >> 5);
    const int lane = threadIdx.x & 31;
    if (e >= E) return;
    const int s = src[e], d = dst[e], r = et[e];
    const float w = norm[e];
    const int f0 = lane << 2;
    const uint2 p = *(const uint2*)(xall + ((size_t)r * N + s) * HIDDEN + f0);
    float* o = out + (size_t)d * HIDDEN + f0;
    atomicAdd(o + 0, bf16_to_f32(p.x & 0xFFFFu) * w);
    atomicAdd(o + 1, bf16_to_f32(p.x >> 16) * w);
    atomicAdd(o + 2, bf16_to_f32(p.y & 0xFFFFu) * w);
    atomicAdd(o + 3, bf16_to_f32(p.y >> 16) * w);
}

// xb = bf16(relu(in)), 4 elems/thread
__global__ void k_relu_bf16(const float* __restrict__ in,
                            ushort_t* __restrict__ outb, int nQ) {
    int i = blockIdx.x * blockDim.x + threadIdx.x;
    if (i >= nQ) return;                      // nQ = N*128/4
    const float4 v = *(const float4*)(in + (size_t)i * 4);
    ushort4 o;
    o.x = f32_to_bf16(fmaxf(v.x, 0.0f)); o.y = f32_to_bf16(fmaxf(v.y, 0.0f));
    o.z = f32_to_bf16(fmaxf(v.z, 0.0f)); o.w = f32_to_bf16(fmaxf(v.w, 0.0f));
    *(ushort4*)(outb + (size_t)i * 4) = o;
}

__global__ void k_relu_inplace(float* __restrict__ io, int nQ) {
    int i = blockIdx.x * blockDim.x + threadIdx.x;
    if (i >= nQ) return;
    float4 v = *(const float4*)(io + (size_t)i * 4);
    v.x = fmaxf(v.x, 0.0f); v.y = fmaxf(v.y, 0.0f);
    v.z = fmaxf(v.z, 0.0f); v.w = fmaxf(v.w, 0.0f);
    *(float4*)(io + (size_t)i * 4) = v;
}

extern "C" void kernel_launch(void* const* d_in, const int* in_sizes, int n_in,
                              void* d_out, int out_size, void* d_ws, size_t ws_size,
                              hipStream_t stream) {
    const int* node_idx = (const int*)d_in[0];
    const int* edge_idx = (const int*)d_in[1];     // [2][E]
    const int* etype    = (const int*)d_in[2];     // [E]
    const float* emb    = (const float*)d_in[3];
    const float* wrel0  = (const float*)d_in[4];
    const float* wroot0 = (const float*)d_in[5];
    const float* b0     = (const float*)d_in[6];
    const float* wrel1  = (const float*)d_in[7];
    const float* wroot1 = (const float*)d_in[8];
    const float* b1     = (const float*)d_in[9];

    const int N = in_sizes[0];
    const int E = in_sizes[2];
    const int* esrc = edge_idx;
    const int* edst = edge_idx + E;

    // ---- workspace carve (256B aligned) ----
    size_t off = 0;
    auto carve = [&](size_t bytes) {
        void* p = (char*)d_ws + off;
        off = (off + bytes + 255) & ~(size_t)255;
        return p;
    };
    ushort_t* xb   = (ushort_t*)carve((size_t)N * HIDDEN * 2);
    ushort_t* wt   = (ushort_t*)carve((size_t)NMAT * HIDDEN * HIDDEN * 2);
    int*      deg  = (int*)carve((size_t)NREL * N * 4);
    float*    norm = (float*)carve((size_t)E * 4);
    ushort_t* xall = (ushort_t*)carve((size_t)NREL * N * HIDDEN * 2);
    float*    o1   = (float*)carve((size_t)N * HIDDEN * 4);
    (void)ws_size;

    const int B = 256;
    const int nQ = N * HIDDEN / 4;            // quad elements
    const int castQ = N * 32;                 // 4 features per thread
    const int rowTiles = (N + 15) / 16;
    dim3 gemmGrid(rowTiles, NMAT);
    const unsigned int scatBlocks = (unsigned int)((E + 7) / 8);  // 8 edges/block

    // ---- edge normalization (shared by both layers) ----
    k_zero_i32<<<(NREL * N + B - 1) / B, B, 0, stream>>>(deg, NREL * N);
    k_count_deg<<<(E + B - 1) / B, B, 0, stream>>>(edst, etype, deg, E, N);
    k_edge_norm<<<(E + B - 1) / B, B, 0, stream>>>(edst, etype, deg, norm, E, N);

    // ---- layer 0 ----
    k_cast_x<<<(castQ + B - 1) / B, B, 0, stream>>>(node_idx, emb, xb, castQ);
    k_prep_w<<<(NMAT * HIDDEN * HIDDEN + B - 1) / B, B, 0, stream>>>(wrel0, wroot0, wt);
    k_gemm<<<gemmGrid, B, 0, stream>>>(xb, wt, b0, xall, o1, N);
    k_scatter<<<scatBlocks, B, 0, stream>>>(xall, esrc, edst, etype, norm, o1, E, N);
    k_relu_bf16<<<(nQ + B - 1) / B, B, 0, stream>>>(o1, xb, nQ);

    // ---- layer 1 ----
    float* out = (float*)d_out;
    k_prep_w<<<(NMAT * HIDDEN * HIDDEN + B - 1) / B, B, 0, stream>>>(wrel1, wroot1, wt);
    k_gemm<<<gemmGrid, B, 0, stream>>>(xb, wt, b1, xall, out, N);
    k_scatter<<<scatBlocks, B, 0, stream>>>(xall, esrc, edst, etype, norm, out, E, N);
    k_relu_inplace<<<(nQ + B - 1) / B, B, 0, stream>>>(out, nQ);
}
